// MultiHeadSelfAttention_50130858279056
// MI455X (gfx1250) — matrix operand order repaired and checked by reading, not compiled, not run on hardware
//
#include <hip/hip_runtime.h>
#include <hip/hip_bf16.h>

// ---------------------------------------------------------------------------
// MHSA forward for MI455X (gfx1250, wave32, WMMA).
// Pipeline: f32->bf16 cvt -> bf16 WMMA GEMM (qkv) -> flash attention (WMMA)
//           -> bf16 WMMA GEMM (+bias, f32 out).
// Double-buffered GEMM K-loop + async global->LDS staging
// (GLOBAL_LOAD_ASYNC_TO_LDS_B128 / ASYNCcnt) where copies are transpose-free.
// ---------------------------------------------------------------------------

typedef __attribute__((ext_vector_type(16))) __bf16         v16bf;
typedef __attribute__((ext_vector_type(8)))  float          v8f;
typedef __attribute__((ext_vector_type(4)))  unsigned int   u32x4;
typedef __attribute__((ext_vector_type(4)))  int            int4v;
typedef __attribute__((ext_vector_type(4)))  float          f32x4;
typedef __attribute__((ext_vector_type(4)))  unsigned short u16x4;

#define AS_GLOBAL __attribute__((address_space(1)))
#define AS_SHARED __attribute__((address_space(3)))

#if defined(__has_builtin)
#  if __has_builtin(__builtin_amdgcn_global_load_async_to_lds_b128)
#    define HAVE_ASYNC_LDS 1
#  endif
#endif
#ifndef HAVE_ASYNC_LDS
#  define HAVE_ASYNC_LDS 0
#endif

__device__ __forceinline__ void async_copy16(const unsigned short* gp,
                                             unsigned short* lp) {
#if HAVE_ASYNC_LDS
  // signature (per hipcc diagnostic): (int4 addrspace(1)*, int4 addrspace(3)*,
  //                                    imm offset, imm cpol)
  __builtin_amdgcn_global_load_async_to_lds_b128(
      (AS_GLOBAL int4v*)gp, (AS_SHARED int4v*)lp, 0, 0);
#else
  *reinterpret_cast<u32x4*>(lp) = *reinterpret_cast<const u32x4*>(gp);
#endif
}

__device__ __forceinline__ void wait_async0() {
#if HAVE_ASYNC_LDS
#  if __has_builtin(__builtin_amdgcn_s_wait_asynccnt)
  __builtin_amdgcn_s_wait_asynccnt(0);
#  else
  asm volatile("s_wait_asynccnt 0x0" ::: "memory");
#  endif
#endif
}

__device__ __forceinline__ unsigned short f2bf_raw(float f) {
  union { float f; unsigned int u; } c; c.f = f;
  unsigned int u = c.u;
  unsigned int r = u + 0x7FFFu + ((u >> 16) & 1u);   // round-to-nearest-even
  return (unsigned short)(r >> 16);
}

union FragU { u32x4 u[2]; v16bf v; };

// Load a 16-element bf16 fragment as two 16B chunks at (elem) offsets off0/off1.
__device__ __forceinline__ v16bf ld_frag(const unsigned short* p, int off0, int off1) {
  FragU c;
  c.u[0] = *reinterpret_cast<const u32x4*>(p + off0);
  c.u[1] = *reinterpret_cast<const u32x4*>(p + off1);
  return c.v;
}

__device__ __forceinline__ v8f wmma_bf16(v16bf a, v16bf b, v8f c) {
  // emits v_wmma_f32_16x16x32_bf16
  return __builtin_amdgcn_wmma_f32_16x16x32_bf16(false, a, false, b, (short)0, c,
                                                 false, false);
}

// ---------------------------------------------------------------------------
// Stage 0: fp32 -> bf16 (vectorized x4)
// ---------------------------------------------------------------------------
__global__ __launch_bounds__(256) void cvt_f32_bf16_kernel(
    const float* __restrict__ in, unsigned short* __restrict__ out, int n4) {
  int i = blockIdx.x * blockDim.x + threadIdx.x;
  if (i < n4) {
    f32x4 f = reinterpret_cast<const f32x4*>(in)[i];
    u16x4 o;
    o.x = f2bf_raw(f.x); o.y = f2bf_raw(f.y);
    o.z = f2bf_raw(f.z); o.w = f2bf_raw(f.w);
    reinterpret_cast<u16x4*>(out)[i] = o;
  }
}

// ---------------------------------------------------------------------------
// Stage 1/3: C = A(bf16)[M,K] x B(bf16)[K,N] + bias, out bf16 or f32.
// 128x128 block tile, 8 waves (2x4), each wave 64x32 (4x2 WMMA tiles), Ktile=32.
// Double-buffered LDS; A tile staged with async global->LDS copies.
// ---------------------------------------------------------------------------
#define GT_M 128
#define GT_N 128
#define GT_K 32
#define A_PITCH 40   // halves; 80B rows, 16B aligned
#define B_PITCH 40

template<int OUT_BF16>
__global__ __launch_bounds__(256) void gemm_bf16_kernel(
    const unsigned short* __restrict__ A,   // [M,K] row major
    const unsigned short* __restrict__ B,   // [K,N] row major
    const float* __restrict__ bias,         // [N]
    void* __restrict__ Cout,                // [M,N]
    int M, int N, int K) {
  __shared__ unsigned short As[2][GT_M * A_PITCH];    // A tiles, row major
  __shared__ unsigned short Bt[2][GT_N * B_PITCH];    // B tiles, transposed [n][k]

  const int tid  = threadIdx.x;
  const int wave = tid >> 5;
  const int lane = tid & 31;
  const int lrow = lane & 15;
  const int hi   = lane >> 4;                 // lane half selector
  const int mBlk = blockIdx.y * GT_M;
  const int nBlk = blockIdx.x * GT_N;
  const int waveM = (wave >> 2) * 64;         // 0 / 64
  const int waveN = (wave & 3) * 32;          // 0..96
  const int aOff = hi ? 16 : 0;               // A-frag K offset per lane half
  const int bOff = hi ? 16 : 0;               // B-frag K offset per lane half

  v8f acc[4][2];
#pragma unroll
  for (int i = 0; i < 4; ++i)
#pragma unroll
    for (int j = 0; j < 2; ++j) acc[i][j] = v8f{};

  // ---- tile staging helpers -------------------------------------------------
  auto stageA = [&](int buf, int k0) {
    // 128x32 halves = 512 chunks of 8 halves, 2 chunks/thread, async b128
#pragma unroll
    for (int i = 0; i < 2; ++i) {
      int c  = tid + i * 256;
      int r  = c >> 2;              // 4 chunks per row
      int cc = (c & 3) * 8;
      async_copy16(A + (size_t)(mBlk + r) * K + k0 + cc,
                   &As[buf][r * A_PITCH + cc]);
    }
  };
  auto stageB = [&](int buf, int k0) {
    // 32(k) x 128(n), transposed into [n][k]: pairs along n (sync path)
#pragma unroll
    for (int i = 0; i < 8; ++i) {
      int p  = tid + i * 256;       // 2048 pairs
      int kk = p >> 6;              // 64 pairs per k-row
      int n2 = (p & 63) * 2;
      unsigned int d = *reinterpret_cast<const unsigned int*>(
          B + (size_t)(k0 + kk) * N + nBlk + n2);
      Bt[buf][(n2    ) * B_PITCH + kk] = (unsigned short)(d & 0xFFFFu);
      Bt[buf][(n2 + 1) * B_PITCH + kk] = (unsigned short)(d >> 16);
    }
  };

  // ---- software pipeline: prologue ------------------------------------------
  stageA(0, 0);
  stageB(0, 0);
  wait_async0();
  __syncthreads();

  int cur = 0;
  for (int k0 = 0; k0 < K; k0 += GT_K) {
    const int nxt = cur ^ 1;
    // issue next tile's loads while current tile is consumed
    if (k0 + GT_K < K) {
      stageA(nxt, k0 + GT_K);
      stageB(nxt, k0 + GT_K);
    }

    v16bf af[4], bfv[2];
#pragma unroll
    for (int im = 0; im < 4; ++im) {
      int r = waveM + im * 16 + lrow;
      af[im] = ld_frag(&As[cur][r * A_PITCH], aOff, aOff + 8);
    }
#pragma unroll
    for (int in = 0; in < 2; ++in) {
      int cidx = waveN + in * 16 + lrow;
      bfv[in] = ld_frag(&Bt[cur][cidx * B_PITCH], bOff, bOff + 8);
    }
#pragma unroll
    for (int im = 0; im < 4; ++im)
#pragma unroll
      for (int in = 0; in < 2; ++in)
        acc[im][in] = wmma_bf16(af[im], bfv[in], acc[im][in]);

    wait_async0();       // next tile's async copies landed (this wave)
    __syncthreads();     // all waves' staging visible
    cur = nxt;
  }

  // ---- epilogue: C-frag layout: lane col = tile_n + lrow, rows = hi*8 + r
#pragma unroll
  for (int in = 0; in < 2; ++in) {
    int colG = nBlk + waveN + in * 16 + lrow;
    float bv = bias[colG];
#pragma unroll
    for (int im = 0; im < 4; ++im) {
      int rowBase = mBlk + waveM + im * 16 + hi * 8;
#pragma unroll
      for (int r = 0; r < 8; ++r) {
        float v = acc[im][in][r] + bv;
        size_t idx = (size_t)(rowBase + r) * N + colG;
        if (OUT_BF16) reinterpret_cast<unsigned short*>(Cout)[idx] = f2bf_raw(v);
        else          reinterpret_cast<float*>(Cout)[idx] = v;
      }
    }
  }
}

// ---------------------------------------------------------------------------
// Stage 2: flash attention. qkv bf16 [B,T,3C] -> attn bf16 [B,T,C].
// Grid: (T/128, H, B); 256 threads = 8 waves; each wave owns 16 query rows.
// K tile kept row-major [key][d] (== B-frag order for Q·K^T), staged async.
// V tile stored transposed [d][key] (== B-frag order for P·V).
// P round-trips through per-wave LDS tile for C->A relayout.
// ---------------------------------------------------------------------------
#define SEQ_T 2048
#define DIM_C 1024
#define NHEAD 16
#define DHEAD 64
#define QBLK  128
#define KBLK  64
#define KV_P  72    // pitch in halves; 144B rows, 16B aligned

__global__ __launch_bounds__(256) void flash_attn_kernel(
    const unsigned short* __restrict__ qkv,   // [B,T,3C] bf16
    unsigned short* __restrict__ attn_out) {  // [B,T,C]  bf16
  __shared__ unsigned short kls[KBLK * KV_P];          // K block [key][d]
  __shared__ unsigned short vls[DHEAD * KV_P];         // V block transposed [d][key]
  __shared__ unsigned short pls[8 * 16 * KV_P];        // per-wave P tiles [16][64]

  const int tid  = threadIdx.x;
  const int wave = tid >> 5;
  const int lane = tid & 31;
  const int lrow = lane & 15;
  const int hi   = lane >> 4;
  const int qb   = blockIdx.x * QBLK;
  const int h    = blockIdx.y;
  const int b    = blockIdx.z;
  const int aOff = hi ? 16 : 0;
  const int bOff = hi ? 16 : 0;
  const float scale = 0.125f;                 // 1/sqrt(64)

  const size_t rowStride = 3 * DIM_C;
  const unsigned short* qbase = qkv + (size_t)b * SEQ_T * rowStride;
  const int qrow0 = qb + wave * 16;

  // Q fragments live in registers for the whole kernel (16 rows x 64 d).
  v16bf qf[2];
  {
    const unsigned short* qp = qbase + (size_t)(qrow0 + lrow) * rowStride + h * DHEAD;
#pragma unroll
    for (int s = 0; s < 2; ++s)
      qf[s] = ld_frag(qp, 32 * s + aOff, 32 * s + aOff + 8);
  }

  v8f oacc[4];
#pragma unroll
  for (int f = 0; f < 4; ++f) oacc[f] = v8f{};
  float rmax[8], rsum[8];
#pragma unroll
  for (int r = 0; r < 8; ++r) { rmax[r] = -3.0e38f; rsum[r] = 0.f; }

  const int jbMax = (qb + QBLK - 1) / KBLK;   // causal: only blocks <= query block
  for (int jb = 0; jb <= jbMax; ++jb) {
    const int j0 = jb * KBLK;
    // ---- stage K block (async): 64x64 halves = 512 chunks of 8
#pragma unroll
    for (int i = 0; i < 2; ++i) {
      int c   = tid + i * 256;
      int key = c >> 3;
      int co  = (c & 7) * 8;
      async_copy16(qbase + (size_t)(j0 + key) * rowStride + DIM_C + h * DHEAD + co,
                   kls + key * KV_P + co);
    }
    // ---- stage V block transposed (sync): pairs along d
#pragma unroll
    for (int i = 0; i < 8; ++i) {
      int p   = tid + i * 256;          // 2048 pairs = 64 keys * 32 d-pairs
      int key = p >> 5;
      int d2  = (p & 31) * 2;
      unsigned int dv = *reinterpret_cast<const unsigned int*>(
          qbase + (size_t)(j0 + key) * rowStride + 2 * DIM_C + h * DHEAD + d2);
      vls[(d2    ) * KV_P + key] = (unsigned short)(dv & 0xFFFFu);
      vls[(d2 + 1) * KV_P + key] = (unsigned short)(dv >> 16);
    }
    wait_async0();
    __syncthreads();

    if (j0 <= qrow0 + 15) {             // wave-uniform: skip fully-masked blocks
      // ---- S = Q K^T  (16x64 per wave, 4 n-tiles, 2 k-steps over d)
      v8f sacc[4];
#pragma unroll
      for (int n = 0; n < 4; ++n) sacc[n] = v8f{};
#pragma unroll
      for (int s = 0; s < 2; ++s)
#pragma unroll
        for (int n = 0; n < 4; ++n) {
          v16bf kf = ld_frag(kls + (n * 16 + lrow) * KV_P,
                             32 * s + bOff, 32 * s + bOff + 8);
          sacc[n] = wmma_bf16(qf[s], kf, sacc[n]);
        }

      // ---- scale + causal mask + online softmax (rows live in lane halves)
      float mnew[8];
#pragma unroll
      for (int r = 0; r < 8; ++r) {
        int qr = qrow0 + hi * 8 + r;
        float mx = -3.0e38f;
#pragma unroll
        for (int n = 0; n < 4; ++n) {
          int col = j0 + n * 16 + lrow;
          float v = sacc[n][r] * scale;
          v = (col <= qr) ? v : -3.0e38f;
          sacc[n][r] = v;
          mx = fmaxf(mx, v);
        }
#pragma unroll
        for (int d = 1; d < 16; d <<= 1)
          mx = fmaxf(mx, __shfl_xor(mx, d, 32));
        mnew[r] = fmaxf(rmax[r], mx);
      }
#pragma unroll
      for (int r = 0; r < 8; ++r) {
        float alpha = __expf(rmax[r] - mnew[r]);
        float sum = 0.f;
#pragma unroll
        for (int n = 0; n < 4; ++n) {
          float p = (sacc[n][r] > -1.0e37f) ? __expf(sacc[n][r] - mnew[r]) : 0.f;
          sacc[n][r] = p;
          sum += p;
        }
#pragma unroll
        for (int d = 1; d < 16; d <<= 1)
          sum += __shfl_xor(sum, d, 32);
        rsum[r] = rsum[r] * alpha + sum;
        rmax[r] = mnew[r];
#pragma unroll
        for (int f = 0; f < 4; ++f) oacc[f][r] *= alpha;
      }

      // ---- P: C-layout -> LDS (bf16) -> A-layout
      unsigned short* pw = pls + wave * 16 * KV_P;
#pragma unroll
      for (int n = 0; n < 4; ++n) {
        int col = n * 16 + lrow;
#pragma unroll
        for (int r = 0; r < 8; ++r)
          pw[(hi * 8 + r) * KV_P + col] = f2bf_raw(sacc[n][r]);
      }
      asm volatile("s_wait_dscnt 0" ::: "memory");   // P store->load turnaround

      // ---- O += P V  (contract over 64 keys: 2 k-steps; 4 d-tiles)
#pragma unroll
      for (int s = 0; s < 2; ++s) {
        v16bf pf = ld_frag(pw + lrow * KV_P, 32 * s + aOff, 32 * s + aOff + 8);
#pragma unroll
        for (int f = 0; f < 4; ++f) {
          v16bf vf = ld_frag(vls + (f * 16 + lrow) * KV_P,
                             32 * s + bOff, 32 * s + bOff + 8);
          oacc[f] = wmma_bf16(pf, vf, oacc[f]);
        }
      }
    }
    __syncthreads();
  }

  // ---- normalize + store bf16 [B,T,C]
  unsigned short* ob = attn_out + (size_t)b * SEQ_T * DIM_C;
#pragma unroll
  for (int r = 0; r < 8; ++r) {
    float inv = 1.0f / rsum[r];
    int rowG = qrow0 + hi * 8 + r;
#pragma unroll
    for (int f = 0; f < 4; ++f) {
      int colG = h * DHEAD + f * 16 + lrow;
      ob[(size_t)rowG * DIM_C + colG] = f2bf_raw(oacc[f][r] * inv);
    }
  }
}

// ---------------------------------------------------------------------------
// Launcher
// ---------------------------------------------------------------------------
extern "C" void kernel_launch(void* const* d_in, const int* in_sizes, int n_in,
                              void* d_out, int out_size, void* d_ws, size_t ws_size,
                              hipStream_t stream) {
  (void)in_sizes; (void)n_in; (void)out_size; (void)ws_size;
  constexpr int B = 4, T = 2048, C = 1024;
  constexpr size_t BT = (size_t)B * T;           // 8192 rows

  const float* x    = (const float*)d_in[0];
  // d_in[1] = attn_mask (causal tril) -- enforced analytically in the kernel
  const float* Wqkv = (const float*)d_in[2];
  const float* bqkv = (const float*)d_in[3];
  const float* Wout = (const float*)d_in[4];
  const float* bout = (const float*)d_in[5];
  float* out = (float*)d_out;

  // workspace layout (bf16 buffers), ~88 MB total
  char* ws = (char*)d_ws;
  unsigned short* xb    = (unsigned short*)ws;                 ws += BT * C * 2;            // 16 MB
  unsigned short* wqkvb = (unsigned short*)ws;                 ws += (size_t)C * 3 * C * 2; // 6 MB
  unsigned short* woutb = (unsigned short*)ws;                 ws += (size_t)C * C * 2;     // 2 MB
  unsigned short* qkvb  = (unsigned short*)ws;                 ws += BT * 3 * C * 2;        // 48 MB
  unsigned short* attnb = (unsigned short*)ws;                                              // 16 MB

  // Stage 0: conversions
  {
    int n4;
    n4 = (int)(BT * C / 4);
    cvt_f32_bf16_kernel<<<(n4 + 255) / 256, 256, 0, stream>>>(x, xb, n4);
    n4 = (int)((size_t)C * 3 * C / 4);
    cvt_f32_bf16_kernel<<<(n4 + 255) / 256, 256, 0, stream>>>(Wqkv, wqkvb, n4);
    n4 = (int)((size_t)C * C / 4);
    cvt_f32_bf16_kernel<<<(n4 + 255) / 256, 256, 0, stream>>>(Wout, woutb, n4);
  }

  // Stage 1: qkv = x @ Wqkv + bqkv   (8192 x 3072 x 1024), bf16 out
  gemm_bf16_kernel<1><<<dim3(3 * C / GT_N, BT / GT_M), 256, 0, stream>>>(
      xb, wqkvb, bqkv, qkvb, (int)BT, 3 * C, C);

  // Stage 2: flash attention -> attnb bf16 [B,T,C]
  flash_attn_kernel<<<dim3(T / QBLK, NHEAD, B), 256, 0, stream>>>(qkvb, attnb);

  // Stage 3: out = attn @ Wout + bout  (8192 x 1024 x 1024), f32 out
  gemm_bf16_kernel<0><<<dim3(C / GT_N, BT / GT_M), 256, 0, stream>>>(
      attnb, woutb, bout, out, (int)BT, C, C);
}
